// NeuralCDE_63866163691999
// MI455X (gfx1250) — matrix operand order repaired and checked by reading, not compiled, not run on hardware
//
#include <hip/hip_runtime.h>
#include <hip/hip_bf16.h>

#define B_   512
#define T_   128
#define C_   16
#define H_   128
#define FH_  128
#define O_   8
#define NCOL (C_ * H_)   // 2048, GEMM2 N dimension
#define AST  136         // padded LDS row stride (halves) -> 272B rows, 16B-aligned sublanes
#define HTS  24          // hid_t row stride (halves): 48B rows, 16B-aligned, low-conflict
#define NT_  512         // scan kernel block size (16 wave32s)

typedef __attribute__((ext_vector_type(16))) _Float16 v16h;
typedef __attribute__((ext_vector_type(8)))  _Float16 v8h;
typedef __attribute__((ext_vector_type(8)))  float    v8f;

// tanh(x) = 1 - 2/(1+e^{2x}) : 5 VALU (2 TRANS), stable at both tails.
__device__ __forceinline__ float fast_tanh(float x) {
  float e = __builtin_amdgcn_exp2f(x * 2.88539008177793f);  // 2*log2(e)
  return 1.f - 2.f * __builtin_amdgcn_rcpf(1.f + e);
}

// ---- fragment loaders (layouts per CDNA5 ISA 7.12.2) ----
__device__ __forceinline__ v16h load_frag_a(const _Float16* p) {
  // halves 0..7 <- p[0..7], halves 8..15 <- p[16..23]  (A 16x32 f16 layout)
  v8h lo = *(const v8h*)(p);
  v8h hi = *(const v8h*)(p + 16);
  return __builtin_shufflevector(lo, hi, 0,1,2,3,4,5,6,7,8,9,10,11,12,13,14,15);
}
__device__ __forceinline__ v16h load_frag_b(const _Float16* p) {
  // 16 contiguous halves: halves 0..15 <- p[0..15]
  v8h lo = *(const v8h*)(p);
  v8h hi = *(const v8h*)(p + 8);
  return __builtin_shufflevector(lo, hi, 0,1,2,3,4,5,6,7,8,9,10,11,12,13,14,15);
}

// ---- kernel 1: Wf2 f32 -> f16, transposed to [h][c][k] (4KB contiguous per h-tile) ----
__global__ __launch_bounds__(256) void wf2_transpose_f16(const float* __restrict__ wf2,
                                                         _Float16* __restrict__ out) {
  int i = blockIdx.x * 256 + threadIdx.x;       // i = hc*128 + k
  if (i < FH_ * NCOL) {
    int k  = i & 127;
    int hc = i >> 7;                            // h*16 + c
    out[i] = (_Float16)wf2[(size_t)k * NCOL + hc];
  }
}

// ---- kernel 2: the RK4 CDE scan, WMMA f16 inputs / f32 accumulate ----
__global__ __launch_bounds__(NT_) void cde_scan(
    const float* __restrict__ coeffs, const float* __restrict__ times,
    const float* __restrict__ Wi1, const float* __restrict__ bi1,
    const float* __restrict__ Wi2, const float* __restrict__ bi2,
    const float* __restrict__ Wf1, const float* __restrict__ bf1,
    const _Float16* __restrict__ Wf2t, const float* __restrict__ bf2,
    float* __restrict__ z_out) {
  __shared__ __align__(16) _Float16 W1z_s[H_ * AST];   // Wf1[1:,:] f16 (GEMM1 B)
  __shared__ __align__(16) _Float16 a_stage[16 * AST]; // staged zz (GEMM1 A)
  __shared__ __align__(16) _Float16 hid_t[FH_ * HTS];  // hid^T (GEMM2 B), 6KB
  __shared__ __align__(16) float kacc[16 * H_];        // stage k output (16 x 128)
  __shared__ __align__(16) float dx_s[16 * C_];        // dX/dt tile
  __shared__ float bias1_s[FH_];
  __shared__ float bf1_s[FH_], w1t_s[FH_];
  __shared__ __align__(16) float bf2_s[NCOL];          // 8KB: GEMM2 bias cache

  const int tid   = threadIdx.x;
  const int lane  = tid & 31;
  const int wave  = tid >> 5;          // 0..15
  const int b0    = blockIdx.x * 16;   // 32 WGs, 16 batch rows each
  const int m_own = tid & 15;          // owned batch row
  const int hg    = tid >> 4;          // owned h-group (4 h's), 0..31
  const int am    = lane & 15;         // A-frag M row
  const int abase = (lane < 16) ? 0 : 8;
  const int cl    = lane & 15;         // C/D frag N (lane) index
  const int mb    = (lane < 16) ? 0 : 8;

  // ---- one-time: weights into LDS ----
  for (int idx = tid; idx < H_ * FH_; idx += NT_) {
    int k = idx >> 7, n = idx & 127;
    W1z_s[n * AST + k] = (_Float16)Wf1[(k + 1) * FH_ + n];
  }
  if (tid < FH_) { bf1_s[tid] = bf1[tid]; w1t_s[tid] = Wf1[tid]; }
  for (int idx = tid; idx < NCOL; idx += NT_) bf2_s[idx] = bf2[idx];

  // ---- z0 = tanh(relu(X0 @ Wi1 + bi1) @ Wi2 + bi2) (scalar, one-time) ----
  float z_reg[4], acc_reg[4];
  {
    float hid20[20];
    const float* x0 = coeffs + (size_t)(b0 + m_own) * T_ * C_;
    for (int j = 0; j < 20; ++j) {
      float s = bi1[j];
      for (int c = 0; c < C_; ++c) s += x0[c] * Wi1[c * 20 + j];
      hid20[j] = fmaxf(s, 0.f);
    }
    for (int i = 0; i < 4; ++i) {
      int h = hg * 4 + i;
      float s = bi2[h];
      for (int j = 0; j < 20; ++j) s += hid20[j] * Wi2[j * H_ + h];
      z_reg[i] = fast_tanh(s);
      acc_reg[i] = 0.f;
      z_out[(size_t)(b0 + m_own) * T_ * H_ + h] = z_reg[i];
    }
  }
  __syncthreads();

  // ---- sequential RK4 scan ----
#pragma unroll 1
  for (int t = 0; t < T_ - 1; ++t) {
    float t0 = times[t];
    float dt = times[t + 1] - t0;
    float rdt = 1.f / dt;                // one uniform divide per step
    if (tid < 256) {  // dX/dt tile: 16x16 elements
      int m = tid >> 4, c = tid & 15;
      const float* cp = coeffs + (size_t)(b0 + m) * T_ * C_ + t * C_ + c;
      dx_s[m * C_ + c] = (cp[C_] - cp[0]) * rdt;
    }
    __syncthreads();
    // per-lane dx for the transposed D' epilogue: dx[m=cl][c=mb+r], contiguous in r
    float dx_reg[8];
    for (int r = 0; r < 8; ++r) dx_reg[r] = dx_s[cl * C_ + mb + r];

#pragma unroll 1
    for (int s = 0; s < 4; ++s) {
      const float alpha = (s == 0) ? 0.f : ((s == 3) ? dt : 0.5f * dt);
      const float wgt   = (s == 1 || s == 2) ? 2.f : 1.f;
      const float teval = t0 + alpha;

      if (tid < FH_) bias1_s[tid] = bf1_s[tid] + teval * w1t_s[tid];
      // stage zz = z + alpha * k_prev  (f16 A matrix for GEMM1)
      for (int i = 0; i < 4; ++i) {
        int h = hg * 4 + i;
        float zz = z_reg[i] + ((s == 0) ? 0.f : alpha * kacc[m_own * H_ + h]);
        a_stage[m_own * AST + h] = (_Float16)zz;
      }
      __syncthreads();

      if (wave < 8) {   // GEMM1: hid(:,16w..16w+15) = relu(zz @ W1z + bias1) -> hid^T
        v16h a1[4];
        for (int kc = 0; kc < 4; ++kc)
          a1[kc] = load_frag_a(&a_stage[am * AST + kc * 32 + abase]);
        const int n0 = wave * 16;
        // bias folded into the WMMA C operand (lane-constant across rows)
        const float bv = bias1_s[n0 + cl];
        v8f acc;
        for (int r = 0; r < 8; ++r) acc[r] = bv;
        for (int kc = 0; kc < 4; ++kc) {
          v16h bfr = load_frag_a(&W1z_s[(n0 + cl) * AST + kc * 32 + abase]);
          acc = __builtin_amdgcn_wmma_f32_16x16x32_f16(false, a1[kc], false, bfr,
                                                       (short)0, acc, false, false);
        }
        v8h pack;
        for (int r = 0; r < 8; ++r)
          pack[r] = (_Float16)fmaxf(acc[r], 0.f);
        // hid^T[k = n0+cl][m = mb .. mb+7]: one aligned b128 store
        *(v8h*)&hid_t[(n0 + cl) * HTS + mb] = pack;
      } else {
        // idle half: pull this wave's first Wf2t tile pair (8KB contiguous) toward WGP
        const _Float16* pf = Wf2t + (size_t)(wave * 8) * 16 * FH_;
        __builtin_prefetch(pf + lane * 64, 0, 3);
        __builtin_prefetch(pf + 2048 + lane * 64, 0, 3);
      }
      __syncthreads();  // hid^T ready

      // GEMM2 (transposed): D'[c,m] = Wf2t_tile(16c x 128k) @ hid^T(128k x 16m) + bf2
      // B fragments (hid^T) are tile-invariant: load once, reuse for all 8 tiles
      v16h b2[4];
      for (int kc = 0; kc < 4; ++kc)
        for (int e = 0; e < 16; ++e)
          b2[kc][e] = hid_t[(kc * 32 + abase + (e & 7) + ((e >> 3) << 4)) * HTS + cl];

#pragma unroll 2
      for (int htp = 0; htp < 4; ++htp) {
        const int h0 = wave * 8 + htp * 2;
        {  // prefetch next contiguous 8KB tile pair
          const _Float16* pf = Wf2t + (size_t)(wave * 8 + ((htp + 1) & 3) * 2) * 16 * FH_;
          __builtin_prefetch(pf + lane * 64, 0, 3);
          __builtin_prefetch(pf + 2048 + lane * 64, 0, 3);
        }
        // bias as initial C accumulator: bf2[h, c=mb+r], contiguous b128 LDS loads
        v8f acc_a, acc_b;
        {
          const float* bp2 = &bf2_s[h0 * 16 + mb];
          for (int r = 0; r < 8; ++r) { acc_a[r] = bp2[r]; acc_b[r] = bp2[16 + r]; }
        }
        const _Float16* ta = Wf2t + (size_t)h0 * 16 * FH_ + am * FH_ + abase;
        for (int kc = 0; kc < 4; ++kc) {
          v16h afa = load_frag_a(ta + kc * 32);
          v16h afb = load_frag_a(ta + 16 * FH_ + kc * 32);
          acc_a = __builtin_amdgcn_wmma_f32_16x16x32_f16(false, afa, false, b2[kc],
                                                         (short)0, acc_a, false, false);
          acc_b = __builtin_amdgcn_wmma_f32_16x16x32_f16(false, afb, false, b2[kc],
                                                         (short)0, acc_b, false, false);
        }
        // D' layout: lane = m (= cl), vgpr r = c (= mb+r): in-lane contraction over c
        float pa = 0.f, pb = 0.f;
        for (int r = 0; r < 8; ++r) {
          pa += fast_tanh(acc_a[r]) * dx_reg[r];
          pb += fast_tanh(acc_b[r]) * dx_reg[r];
        }
        pa += __shfl_xor(pa, 16, 32);   // combine the two c-halves
        pb += __shfl_xor(pb, 16, 32);
        if (lane < 16)
          *(float2*)&kacc[cl * H_ + h0] = make_float2(pa, pb);
      }
      __syncthreads();

      for (int i = 0; i < 4; ++i)
        acc_reg[i] += wgt * kacc[m_own * H_ + (hg * 4 + i)];
    }

    // z_{t+1} = z_t + dt/6 * (k1 + 2k2 + 2k3 + k4)
    const float c6 = dt * (1.f / 6.f);
    for (int i = 0; i < 4; ++i) {
      int h = hg * 4 + i;
      z_reg[i] += c6 * acc_reg[i];
      acc_reg[i] = 0.f;
      z_out[(size_t)(b0 + m_own) * T_ * H_ + (size_t)(t + 1) * H_ + h] = z_reg[i];
    }
    __syncthreads();
  }
}

// ---- kernel 3: pred_y = z @ Wr + br  (67 MFLOP, scalar is fine) ----
__global__ __launch_bounds__(256) void readout(const float* __restrict__ z,
                                               const float* __restrict__ Wr,
                                               const float* __restrict__ br,
                                               float* __restrict__ y) {
  int idx = blockIdx.x * 256 + threadIdx.x;
  if (idx >= B_ * T_ * O_) return;
  int o = idx & (O_ - 1);
  int bt = idx >> 3;
  const float* zp = z + (size_t)bt * H_;
  float sacc = br[o];
  for (int h = 0; h < H_; ++h) sacc += zp[h] * Wr[h * O_ + o];
  y[idx] = sacc;
}

extern "C" void kernel_launch(void* const* d_in, const int* in_sizes, int n_in,
                              void* d_out, int out_size, void* d_ws, size_t ws_size,
                              hipStream_t stream) {
  (void)in_sizes; (void)n_in; (void)out_size; (void)ws_size;
  const float* coeffs = (const float*)d_in[0];
  const float* times  = (const float*)d_in[1];
  const float* Wi1    = (const float*)d_in[2];
  const float* bi1    = (const float*)d_in[3];
  const float* Wi2    = (const float*)d_in[4];
  const float* bi2    = (const float*)d_in[5];
  const float* Wf1    = (const float*)d_in[6];
  const float* bf1    = (const float*)d_in[7];
  const float* Wf2    = (const float*)d_in[8];
  const float* bf2    = (const float*)d_in[9];
  const float* Wr     = (const float*)d_in[10];
  const float* br     = (const float*)d_in[11];

  float* z_out = (float*)d_out;                       // (B, T, H)
  float* y_out = z_out + (size_t)B_ * T_ * H_;        // (B, T, O)
  _Float16* Wf2t = (_Float16*)d_ws;                   // 512 KB f16 transposed Wf2

  wf2_transpose_f16<<<(FH_ * NCOL + 255) / 256, 256, 0, stream>>>(Wf2, Wf2t);
  cde_scan<<<B_ / 16, NT_, 0, stream>>>(coeffs, times, Wi1, bi1, Wi2, bi2,
                                        Wf1, bf1, Wf2t, bf2, z_out);
  readout<<<(B_ * T_ * O_ + 255) / 256, 256, 0, stream>>>(z_out, Wr, br, y_out);
}
